// PointTransformer_55362128445860
// MI455X (gfx1250) — matrix operand order repaired and checked by reading, not compiled, not run on hardware
//
#include <hip/hip_runtime.h>
#include <hip/hip_bf16.h>
#include <math.h>

typedef __bf16 bf16;
typedef __attribute__((ext_vector_type(16))) __bf16 v16bf;
typedef __attribute__((ext_vector_type(8)))  __bf16 v8bf;
typedef __attribute__((ext_vector_type(8)))  float  v8f;

#define BATCH 8
#define NPTS  8192
#define NG    512
#define KNN   32
#define CDIM  384
#define DEPTH 12
#define NHEAD 6
#define HDIM  64
#define ENCD  256
#define NTOK  513              // 1 cls + 512 groups
#define TOKTOT (BATCH * NTOK)  // 4104
#define ROWS  (BATCH * NG * KNN) // 131072

static __device__ __forceinline__ int imin(int a, int b) { return a < b ? a : b; }

static __device__ __forceinline__ v8f wmma_bf16(v16bf a, v16bf b, v8f c) {
  return __builtin_amdgcn_wmma_f32_16x16x32_bf16(false, a, false, b, (short)0, c, false, false);
}

static __device__ __forceinline__ float gelu_exact(float x) {
  return 0.5f * x * (1.0f + erff(x * 0.70710678118654752f));
}

// ---------------- weight conversion ----------------
// dst[n*Kc + k] = src[n*ld + k]   (weight already out×in, K contiguous)
__global__ void wconv_nt(const float* __restrict__ src, bf16* __restrict__ dst,
                         int Nr, int Kc, int ld) {
  int i = blockIdx.x * blockDim.x + threadIdx.x;
  int tot = Nr * Kc;
  if (i < tot) {
    int n = i / Kc, k = i % Kc;
    dst[i] = (bf16)src[(size_t)n * ld + k];
  }
}
// src is K×N (N contiguous, x@W layout) -> dst N×K
__global__ void wconv_t(const float* __restrict__ src, bf16* __restrict__ dst,
                        int Kc, int Nc) {
  int i = blockIdx.x * blockDim.x + threadIdx.x;
  int tot = Kc * Nc;
  if (i < tot) {
    int n = i / Kc, k = i % Kc;
    dst[i] = (bf16)src[(size_t)k * Nc + n];
  }
}

__global__ void bn_prep(const float* g, const float* b, const float* m,
                        const float* v, float* sc, float* sh, int n) {
  int i = blockIdx.x * blockDim.x + threadIdx.x;
  if (i < n) {
    float s = g[i] * rsqrtf(v[i] + 1e-5f);
    sc[i] = s;
    sh[i] = b[i] - m[i] * s;
  }
}

// ---------------- FPS ----------------
__global__ __launch_bounds__(1024) void fps_kernel(const float* __restrict__ pts,
                                                   float* __restrict__ center) {
  int b = blockIdx.x, tid = threadIdx.x;
  __shared__ float dist[NPTS];
  __shared__ float rv[1024];
  __shared__ int   ri[1024];
  __shared__ int   sfar;
  const float* pb = pts + (size_t)b * NPTS * 3;
  for (int i = tid; i < NPTS; i += 1024) dist[i] = 1e10f;
  __syncthreads();
  int far = 0;
  for (int g = 0; g < NG; g++) {
    float cx = pb[far * 3 + 0], cy = pb[far * 3 + 1], cz = pb[far * 3 + 2];
    if (tid == 0) {
      size_t o = ((size_t)b * NG + g) * 3;
      center[o + 0] = cx; center[o + 1] = cy; center[o + 2] = cz;
    }
    float lm = -1.f; int li = 0;
    for (int i = tid; i < NPTS; i += 1024) {
      float dx = pb[i * 3] - cx, dy = pb[i * 3 + 1] - cy, dz = pb[i * 3 + 2] - cz;
      float d = dx * dx + dy * dy + dz * dz;
      float nd = fminf(dist[i], d);
      dist[i] = nd;
      if (nd > lm) { lm = nd; li = i; }
    }
    rv[tid] = lm; ri[tid] = li;
    __syncthreads();
    for (int st = 512; st > 0; st >>= 1) {
      if (tid < st) {
        if (rv[tid + st] > rv[tid]) { rv[tid] = rv[tid + st]; ri[tid] = ri[tid + st]; }
      }
      __syncthreads();
    }
    if (tid == 0) sfar = ri[0];
    __syncthreads();
    far = sfar;
  }
}

// ---------------- kNN + gather (centered neighborhoods) ----------------
__global__ __launch_bounds__(256) void knn_kernel(const float* __restrict__ pts,
                                                  const float* __restrict__ center,
                                                  float* __restrict__ nb) {
  int bg = blockIdx.x, tid = threadIdx.x;
  int b = bg >> 9;
  __shared__ float dist[NPTS];
  __shared__ float rv[256];
  __shared__ int   ri[256];
  __shared__ int   sel[KNN];
  const float* pb = pts + (size_t)b * NPTS * 3;
  float cx = center[bg * 3 + 0], cy = center[bg * 3 + 1], cz = center[bg * 3 + 2];
  for (int i = tid; i < NPTS; i += 256) {
    float dx = pb[i * 3] - cx, dy = pb[i * 3 + 1] - cy, dz = pb[i * 3 + 2] - cz;
    dist[i] = dx * dx + dy * dy + dz * dz;
  }
  __syncthreads();
  for (int j = 0; j < KNN; j++) {
    float lm = 3e38f; int li = 0;
    for (int i = tid; i < NPTS; i += 256) {
      float d = dist[i];
      if (d < lm) { lm = d; li = i; }
    }
    rv[tid] = lm; ri[tid] = li;
    __syncthreads();
    for (int st = 128; st > 0; st >>= 1) {
      if (tid < st) {
        if (rv[tid + st] < rv[tid]) { rv[tid] = rv[tid + st]; ri[tid] = ri[tid + st]; }
      }
      __syncthreads();
    }
    if (tid == 0) { sel[j] = ri[0]; dist[ri[0]] = 3e38f; }
    __syncthreads();
  }
  if (tid < KNN * 3) {
    int j = tid / 3, c = tid % 3;
    float cen = (c == 0) ? cx : (c == 1) ? cy : cz;
    nb[((size_t)bg * KNN + j) * 3 + c] = pb[(size_t)sel[j] * 3 + c] - cen;
  }
}

// ---------------- encoder stage 1 (K=3, scalar) ----------------
__global__ __launch_bounds__(128) void enc1_kernel(const float* __restrict__ nb,
                                                   const float* __restrict__ w,
                                                   const float* __restrict__ bia,
                                                   const float* __restrict__ g,
                                                   const float* __restrict__ bb,
                                                   const float* __restrict__ mm,
                                                   const float* __restrict__ vv,
                                                   bf16* __restrict__ x1) {
  int row = blockIdx.x, o = threadIdx.x;
  size_t rb = (size_t)row * 3;
  float s = bia[o] + nb[rb] * w[o * 3] + nb[rb + 1] * w[o * 3 + 1] + nb[rb + 2] * w[o * 3 + 2];
  float sc = g[o] * rsqrtf(vv[o] + 1e-5f);
  float y = (s - mm[o]) * sc + bb[o];
  x1[(size_t)row * 128 + o] = (bf16)fmaxf(y, 0.f);
}

// ---------------- maxpool over neighborhood (KNN) ----------------
__global__ __launch_bounds__(256) void maxpool_kernel(const bf16* __restrict__ in,
                                                      bf16* __restrict__ out, int C) {
  int g = blockIdx.x;
  for (int c = threadIdx.x; c < C; c += 256) {
    float m = -3e38f;
    for (int kk = 0; kk < KNN; kk++)
      m = fmaxf(m, (float)in[((size_t)g * KNN + kk) * C + c]);
    out[(size_t)g * C + c] = (bf16)m;
  }
}

// ---------------- pos embedding stage 1 (K=3, gelu) ----------------
__global__ __launch_bounds__(128) void pos1_kernel(const float* __restrict__ center,
                                                   const float* __restrict__ w,
                                                   const float* __restrict__ b,
                                                   bf16* __restrict__ p) {
  int row = blockIdx.x, o = threadIdx.x;
  size_t rb = (size_t)row * 3;
  float s = b[o] + center[rb] * w[o] + center[rb + 1] * w[128 + o] + center[rb + 2] * w[256 + o];
  p[(size_t)row * 128 + o] = (bf16)gelu_exact(s);
}

// ---------------- generic bf16 WMMA GEMM ----------------
// C[m,n] = sum_k A[m,k]*Bw[n,k]  (A: M×K row-major bf16, Bw: N×K row-major bf16)
// epilogue: +bias[n], +rowbias[(m>>rbshift)*N+n], bn affine, relu/gelu, +resid, out f32 or bf16
__global__ __launch_bounds__(256) void gemm_bf16_kernel(
    const bf16* __restrict__ A, const bf16* __restrict__ Bw,
    float* __restrict__ Cf, bf16* __restrict__ Cb,
    const float* __restrict__ bias, const float* __restrict__ rowbias, int rbshift,
    const float* __restrict__ bnsc, const float* __restrict__ bnsh,
    const float* __restrict__ resid,
    int M, int N, int K, int act) {
  int lane = threadIdx.x & 31;
  int wave = threadIdx.x >> 5;
  int wm = wave & 3, wn = wave >> 2;       // 4×2 waves -> 128×64 block tile
  int m0 = blockIdx.y * 128 + wm * 32;
  int n0 = blockIdx.x * 64 + wn * 32;
  if (m0 >= M || n0 >= N) return;          // no barriers in kernel: safe per-wave exit
  int r = lane & 15, hi = lane >> 4;
  const bf16* ap0 = A + (size_t)imin(m0 + r, M - 1) * K;
  const bf16* ap1 = A + (size_t)imin(m0 + 16 + r, M - 1) * K;
  const bf16* bp0 = Bw + (size_t)imin(n0 + r, N - 1) * K;
  const bf16* bp1 = Bw + (size_t)imin(n0 + 16 + r, N - 1) * K;
  v8f acc00 = {}, acc01 = {}, acc10 = {}, acc11 = {};
  union U { v16bf v; v8bf h[2]; };
  for (int k = 0; k < K; k += 32) {
    U a0, a1, b0, b1;
    a0.h[0] = *(const v8bf*)(ap0 + k + 8 * hi);
    a0.h[1] = *(const v8bf*)(ap0 + k + 16 + 8 * hi);
    a1.h[0] = *(const v8bf*)(ap1 + k + 8 * hi);
    a1.h[1] = *(const v8bf*)(ap1 + k + 16 + 8 * hi);
    b0.h[0] = *(const v8bf*)(bp0 + k + 8 * hi);
    b0.h[1] = *(const v8bf*)(bp0 + k + 16 + 8 * hi);
    b1.h[0] = *(const v8bf*)(bp1 + k + 8 * hi);
    b1.h[1] = *(const v8bf*)(bp1 + k + 16 + 8 * hi);
    __builtin_prefetch((const void*)(ap0 + k + 256), 0, 0);
    __builtin_prefetch((const void*)(bp0 + k + 256), 0, 0);
    acc00 = wmma_bf16(a0.v, b0.v, acc00);
    acc01 = wmma_bf16(a0.v, b1.v, acc01);
    acc10 = wmma_bf16(a1.v, b0.v, acc10);
    acc11 = wmma_bf16(a1.v, b1.v, acc11);
  }
#pragma unroll
  for (int i = 0; i < 2; i++) {
#pragma unroll
    for (int j = 0; j < 2; j++) {
      v8f acc = (i == 0) ? ((j == 0) ? acc00 : acc01) : ((j == 0) ? acc10 : acc11);
#pragma unroll
      for (int rr = 0; rr < 8; rr++) {
        int m = m0 + 16 * i + rr + 8 * hi;
        int n = n0 + 16 * j + r;
        if (m < M && n < N) {
          float x = acc[rr];
          if (bias)    x += bias[n];
          if (rowbias) x += rowbias[(size_t)(m >> rbshift) * N + n];
          if (bnsc)    x = x * bnsc[n] + bnsh[n];
          if (act == 1)      x = fmaxf(x, 0.f);
          else if (act == 2) x = gelu_exact(x);
          if (resid)   x += resid[(size_t)m * N + n];
          if (Cf) Cf[(size_t)m * N + n] = x;
          else    Cb[(size_t)m * N + n] = (bf16)x;
        }
      }
    }
  }
}

// ---------------- token assembly ----------------
__global__ __launch_bounds__(CDIM) void assemble_kernel(const float* __restrict__ tok,
                                                        const float* __restrict__ post,
                                                        const float* __restrict__ cls_t,
                                                        const float* __restrict__ cls_p,
                                                        float* __restrict__ xb,
                                                        float* __restrict__ pb) {
  int t = blockIdx.x, c = threadIdx.x;
  int b = t / NTOK, i = t % NTOK;
  size_t o = (size_t)t * CDIM + c;
  if (i == 0) {
    xb[o] = cls_t[c];
    pb[o] = cls_p[c];
  } else {
    size_t bg = ((size_t)b * NG + (i - 1)) * CDIM + c;
    xb[o] = tok[bg];
    pb[o] = post[bg];
  }
}

// ---------------- (x += pos) + LayerNorm ----------------
__global__ __launch_bounds__(128) void ln_kernel(float* __restrict__ x,
                                                 const float* __restrict__ pos,
                                                 const float* __restrict__ gm,
                                                 const float* __restrict__ bt,
                                                 bf16* __restrict__ obf,
                                                 float* __restrict__ ofl) {
  int row = blockIdx.x, tid = threadIdx.x;
  size_t base = (size_t)row * CDIM;
  __shared__ float red[128];
  float v0[3];
  float s = 0.f;
#pragma unroll
  for (int e = 0; e < 3; e++) {
    int c = tid + 128 * e;
    float t = x[base + c];
    if (pos) t += pos[base + c];
    v0[e] = t;
    s += t;
  }
  if (pos) {
#pragma unroll
    for (int e = 0; e < 3; e++) x[base + tid + 128 * e] = v0[e];
  }
  red[tid] = s;
  __syncthreads();
  for (int st = 64; st > 0; st >>= 1) {
    if (tid < st) red[tid] += red[tid + st];
    __syncthreads();
  }
  float mean = red[0] / (float)CDIM;
  __syncthreads();
  float s2 = 0.f;
#pragma unroll
  for (int e = 0; e < 3; e++) { float d = v0[e] - mean; s2 += d * d; }
  red[tid] = s2;
  __syncthreads();
  for (int st = 64; st > 0; st >>= 1) {
    if (tid < st) red[tid] += red[tid + st];
    __syncthreads();
  }
  float rstd = rsqrtf(red[0] / (float)CDIM + 1e-5f);
#pragma unroll
  for (int e = 0; e < 3; e++) {
    int c = tid + 128 * e;
    float y = (v0[e] - mean) * rstd * gm[c] + bt[c];
    if (obf) obf[base + c] = (bf16)y;
    if (ofl) ofl[base + c] = y;
  }
}

// ---------------- fused attention (per b, head, 16-query tile) ----------------
__global__ __launch_bounds__(128) void attn_kernel(const bf16* __restrict__ qkv,
                                                   bf16* __restrict__ out) {
  int qt = blockIdx.x, h = blockIdx.y, b = blockIdx.z;
  int tid = threadIdx.x, lane = tid & 31, wave = tid >> 5;
  int r = lane & 15, hi = lane >> 4;
  __shared__ float S[16][544];     // scores / probabilities (keys padded to 544)
  __shared__ float red2[16][8];
  const int QO = h * HDIM;
  const int KO = (NHEAD + h) * HDIM;
  const int VO = (2 * NHEAD + h) * HDIM;
  const int LD = 3 * CDIM;         // 1152
  // ---- phase 1: S = (q * HD^-0.5) @ k^T via WMMA, 33 key tiles of 16 ----
  int qrow = b * NTOK + imin(qt * 16 + r, NTOK - 1);
  const bf16* qp = qkv + (size_t)qrow * LD + QO;
  union U { v16bf v; v8bf h2[2]; };
  for (int kt = wave; kt < 33; kt += 4) {
    int krow = b * NTOK + imin(kt * 16 + r, NTOK - 1);
    const bf16* kp = qkv + (size_t)krow * LD + KO;
    v8f acc = {};
#pragma unroll
    for (int d = 0; d < HDIM; d += 32) {
      U a, bb;
      a.h2[0]  = *(const v8bf*)(qp + d + 8 * hi);
      a.h2[1]  = *(const v8bf*)(qp + d + 16 + 8 * hi);
      bb.h2[0] = *(const v8bf*)(kp + d + 8 * hi);
      bb.h2[1] = *(const v8bf*)(kp + d + 16 + 8 * hi);
      acc = wmma_bf16(a.v, bb.v, acc);
    }
#pragma unroll
    for (int rr = 0; rr < 8; rr++) {
      int m = rr + 8 * hi;
      int col = kt * 16 + r;
      S[m][col] = (col < NTOK) ? acc[rr] * 0.125f : -3e38f;  // HD^-0.5 = 1/8
    }
  }
  __syncthreads();
  // ---- phase 2: row softmax in LDS (8 threads / row) ----
  int rrow = tid >> 3, sub = tid & 7;
  float mx = -3e38f;
  for (int c = sub; c < 528; c += 8) mx = fmaxf(mx, S[rrow][c]);
  red2[rrow][sub] = mx;
  __syncthreads();
  if (sub == 0) {
    float m2 = red2[rrow][0];
    for (int e = 1; e < 8; e++) m2 = fmaxf(m2, red2[rrow][e]);
    red2[rrow][0] = m2;
  }
  __syncthreads();
  float rm = red2[rrow][0];
  float sum = 0.f;
  for (int c = sub; c < 528; c += 8) {
    float e = __expf(S[rrow][c] - rm);
    S[rrow][c] = e;
    sum += e;
  }
  __syncthreads();
  red2[rrow][sub] = sum;
  __syncthreads();
  if (sub == 0) {
    float s2 = 0.f;
    for (int e = 0; e < 8; e++) s2 += red2[rrow][e];
    red2[rrow][0] = s2;
  }
  __syncthreads();
  float rs = 1.f / red2[rrow][0];
  for (int c = sub; c < 528; c += 8) S[rrow][c] *= rs;
  for (int c = 528 + sub; c < 544; c += 8) S[rrow][c] = 0.f;
  __syncthreads();
  // ---- phase 3: O = P @ V via WMMA; each wave owns one 16-wide HD tile ----
  int j = wave;  // 0..3
  v8f acc = {};
  int d = VO + j * 16 + r;
  for (int k = 0; k < 544; k += 32) {
    v16bf a;
#pragma unroll
    for (int e = 0; e < 8; e++) {
      a[e]     = (bf16)S[r][k + 8 * hi + e];
      a[e + 8] = (bf16)S[r][k + 16 + 8 * hi + e];
    }
    v16bf bb;
#pragma unroll
    for (int e = 0; e < 16; e++) {
      int key = k + (e < 8 ? 8 * hi + e : 8 + 8 * hi + e);
      int trow = b * NTOK + imin(key, NTOK - 1);  // padded keys have P==0
      bb[e] = qkv[(size_t)trow * LD + d];
    }
    acc = wmma_bf16(a, bb, acc);
  }
#pragma unroll
  for (int rr = 0; rr < 8; rr++) {
    int q = qt * 16 + rr + 8 * hi;
    if (q < NTOK)
      out[((size_t)(b * NTOK + q)) * CDIM + h * HDIM + j * 16 + r] = (bf16)acc[rr];
  }
}

// ---------------- classification head ----------------
__global__ __launch_bounds__(256) void head_kernel(const float* __restrict__ xn,
                                                   const float* __restrict__ w1,
                                                   const float* __restrict__ b1,
                                                   const float* __restrict__ w2,
                                                   const float* __restrict__ b2,
                                                   float* __restrict__ out) {
  int b = blockIdx.x, tid = threadIdx.x;
  __shared__ float feat[2 * CDIM];
  __shared__ float red[256];
  for (int d = tid; d < CDIM; d += 256) {
    feat[d] = xn[((size_t)b * NTOK) * CDIM + d];
    float m = -3e38f;
    for (int i = 1; i < NTOK; i++)
      m = fmaxf(m, xn[((size_t)b * NTOK + i) * CDIM + d]);
    feat[CDIM + d] = m;
  }
  __syncthreads();
  float s = b1[tid];
  for (int k = 0; k < 2 * CDIM; k++) s += feat[k] * w1[(size_t)k * 256 + tid];
  red[tid] = fmaxf(s, 0.f) * w2[tid];
  __syncthreads();
  for (int st = 128; st > 0; st >>= 1) {
    if (tid < st) red[tid] += red[tid + st];
    __syncthreads();
  }
  if (tid == 0) out[b] = 1.f / (1.f + expf(-(red[0] + b2[0])));
}

// =====================================================================
extern "C" void kernel_launch(void* const* d_in, const int* in_sizes, int n_in,
                              void* d_out, int out_size, void* d_ws, size_t ws_size,
                              hipStream_t stream) {
  (void)in_sizes; (void)n_in; (void)out_size; (void)ws_size;
  const float* pts     = (const float*)d_in[0];
  const float* enc1_w  = (const float*)d_in[1];
  const float* enc1_b  = (const float*)d_in[2];
  const float* bn1_g   = (const float*)d_in[3];
  const float* bn1_b   = (const float*)d_in[4];
  const float* bn1_m   = (const float*)d_in[5];
  const float* bn1_v   = (const float*)d_in[6];
  const float* enc2_w  = (const float*)d_in[7];
  const float* enc2_b  = (const float*)d_in[8];
  const float* enc3_w  = (const float*)d_in[9];
  const float* enc3_b  = (const float*)d_in[10];
  const float* bn2_g   = (const float*)d_in[11];
  const float* bn2_b   = (const float*)d_in[12];
  const float* bn2_m   = (const float*)d_in[13];
  const float* bn2_v   = (const float*)d_in[14];
  const float* enc4_w  = (const float*)d_in[15];
  const float* enc4_b  = (const float*)d_in[16];
  const float* red_w   = (const float*)d_in[17];
  const float* red_b   = (const float*)d_in[18];
  const float* cls_tok = (const float*)d_in[19];
  const float* cls_pos = (const float*)d_in[20];
  const float* pos1_w  = (const float*)d_in[21];
  const float* pos1_b  = (const float*)d_in[22];
  const float* pos2_w  = (const float*)d_in[23];
  const float* pos2_b  = (const float*)d_in[24];
  const float* n1_g    = (const float*)d_in[25];
  const float* n1_b    = (const float*)d_in[26];
  const float* qkv_w   = (const float*)d_in[27];
  const float* proj_w  = (const float*)d_in[28];
  const float* proj_b  = (const float*)d_in[29];
  const float* n2_g    = (const float*)d_in[30];
  const float* n2_b    = (const float*)d_in[31];
  const float* fc1_w   = (const float*)d_in[32];
  const float* fc1_b   = (const float*)d_in[33];
  const float* fc2_w   = (const float*)d_in[34];
  const float* fc2_b   = (const float*)d_in[35];
  const float* nf_g    = (const float*)d_in[36];
  const float* nf_b    = (const float*)d_in[37];
  const float* h1_w    = (const float*)d_in[38];
  const float* h1_b    = (const float*)d_in[39];
  const float* h2_w    = (const float*)d_in[40];
  const float* h2_b    = (const float*)d_in[41];

  // ---- workspace layout ----
  char* wsb = (char*)d_ws;
  size_t off = 0;
  auto alloc = [&](size_t bytes) -> void* {
    void* p = wsb + off;
    off += (bytes + 255) & ~(size_t)255;
    return p;
  };
  bf16* w_enc2  = (bf16*)alloc((size_t)256 * 128 * 2);
  bf16* w_enc3a = (bf16*)alloc((size_t)512 * 256 * 2);
  bf16* w_enc3b = (bf16*)alloc((size_t)512 * 256 * 2);
  bf16* w_enc4  = (bf16*)alloc((size_t)256 * 512 * 2);
  bf16* w_red   = (bf16*)alloc((size_t)CDIM * 256 * 2);
  bf16* w_pos2  = (bf16*)alloc((size_t)CDIM * 128 * 2);
  bf16* w_qkv   = (bf16*)alloc((size_t)DEPTH * 1152 * CDIM * 2);
  bf16* w_proj  = (bf16*)alloc((size_t)DEPTH * CDIM * CDIM * 2);
  bf16* w_fc1   = (bf16*)alloc((size_t)DEPTH * 1536 * CDIM * 2);
  bf16* w_fc2   = (bf16*)alloc((size_t)DEPTH * CDIM * 1536 * 2);
  float* bn2sc  = (float*)alloc(512 * 4);
  float* bn2sh  = (float*)alloc(512 * 4);
  float* center = (float*)alloc((size_t)BATCH * NG * 3 * 4);
  float* nb     = (float*)alloc((size_t)ROWS * 3 * 4);
  bf16* x1      = (bf16*)alloc((size_t)ROWS * 128 * 2);
  bf16* x2      = (bf16*)alloc((size_t)ROWS * 256 * 2);
  bf16* gmax    = (bf16*)alloc((size_t)BATCH * NG * 256 * 2);
  float* tbuf   = (float*)alloc((size_t)BATCH * NG * 512 * 4);
  bf16* x3      = (bf16*)alloc((size_t)ROWS * 512 * 2);
  bf16* x4      = (bf16*)alloc((size_t)ROWS * 256 * 2);
  float* tok    = (float*)alloc((size_t)BATCH * NG * CDIM * 4);
  bf16* p128    = (bf16*)alloc((size_t)BATCH * NG * 128 * 2);
  float* post   = (float*)alloc((size_t)BATCH * NG * CDIM * 4);
  float* xbuf   = (float*)alloc((size_t)TOKTOT * CDIM * 4);
  float* posb   = (float*)alloc((size_t)TOKTOT * CDIM * 4);
  bf16* hnb     = (bf16*)alloc((size_t)TOKTOT * CDIM * 2);
  bf16* qkvb    = (bf16*)alloc((size_t)TOKTOT * 1152 * 2);
  bf16* attb    = (bf16*)alloc((size_t)TOKTOT * CDIM * 2);
  bf16* mbuf    = (bf16*)alloc((size_t)TOKTOT * 1536 * 2);
  float* xn     = (float*)alloc((size_t)TOKTOT * CDIM * 4);

  auto cvt_nt = [&](const float* s, bf16* d, int Nr, int Kc, int ld) {
    int tot = Nr * Kc;
    wconv_nt<<<(tot + 255) / 256, 256, 0, stream>>>(s, d, Nr, Kc, ld);
  };
  auto cvt_t = [&](const float* s, bf16* d, int Kc, int Nc) {
    int tot = Kc * Nc;
    wconv_t<<<(tot + 255) / 256, 256, 0, stream>>>(s, d, Kc, Nc);
  };

  // weight conversions
  cvt_nt(enc2_w, w_enc2, 256, 128, 128);
  cvt_nt(enc3_w,       w_enc3a, 512, 256, 512);  // cols 0..255  (gmax part)
  cvt_nt(enc3_w + 256, w_enc3b, 512, 256, 512);  // cols 256..511 (x2 part)
  cvt_nt(enc4_w, w_enc4, 256, 512, 512);
  cvt_t(red_w,  w_red,  256, CDIM);   // (256,384) K×N -> N×K
  cvt_t(pos2_w, w_pos2, 128, CDIM);
  for (int l = 0; l < DEPTH; l++) {
    cvt_t(qkv_w  + (size_t)l * CDIM * 1152, w_qkv  + (size_t)l * 1152 * CDIM, CDIM, 1152);
    cvt_t(proj_w + (size_t)l * CDIM * CDIM, w_proj + (size_t)l * CDIM * CDIM, CDIM, CDIM);
    cvt_t(fc1_w  + (size_t)l * CDIM * 1536, w_fc1  + (size_t)l * 1536 * CDIM, CDIM, 1536);
    cvt_t(fc2_w  + (size_t)l * 1536 * CDIM, w_fc2  + (size_t)l * CDIM * 1536, 1536, CDIM);
  }
  bn_prep<<<2, 256, 0, stream>>>(bn2_g, bn2_b, bn2_m, bn2_v, bn2sc, bn2sh, 512);

  auto gemm = [&](const bf16* A, const bf16* Bw, const float* bias,
                  const float* rowbias, int rbshift, const float* bnsc,
                  const float* bnsh, const float* resid, float* Cf, bf16* Cb,
                  int M, int N, int K, int act) {
    dim3 g((N + 63) / 64, (M + 127) / 128);
    gemm_bf16_kernel<<<g, 256, 0, stream>>>(A, Bw, Cf, Cb, bias, rowbias, rbshift,
                                            bnsc, bnsh, resid, M, N, K, act);
  };

  // ---- grouping ----
  fps_kernel<<<BATCH, 1024, 0, stream>>>(pts, center);
  knn_kernel<<<BATCH * NG, 256, 0, stream>>>(pts, center, nb);

  // ---- encoder ----
  enc1_kernel<<<ROWS, 128, 0, stream>>>(nb, enc1_w, enc1_b, bn1_g, bn1_b, bn1_m, bn1_v, x1);
  gemm(x1, w_enc2, enc2_b, nullptr, 0, nullptr, nullptr, nullptr, nullptr, x2,
       ROWS, 256, 128, 0);
  maxpool_kernel<<<BATCH * NG, 256, 0, stream>>>(x2, gmax, 256);
  gemm(gmax, w_enc3a, enc3_b, nullptr, 0, nullptr, nullptr, nullptr, tbuf, nullptr,
       BATCH * NG, 512, 256, 0);                                  // t = gmax@Wa + b3
  gemm(x2, w_enc3b, nullptr, tbuf, 5, bn2sc, bn2sh, nullptr, nullptr, x3,
       ROWS, 512, 256, 1);                                        // x3 = relu(bn(x2@Wb + t))
  gemm(x3, w_enc4, enc4_b, nullptr, 0, nullptr, nullptr, nullptr, nullptr, x4,
       ROWS, 256, 512, 0);
  maxpool_kernel<<<BATCH * NG, 256, 0, stream>>>(x4, gmax, 256);  // reuse gmax
  gemm(gmax, w_red, red_b, nullptr, 0, nullptr, nullptr, nullptr, tok, nullptr,
       BATCH * NG, CDIM, 256, 0);
  pos1_kernel<<<BATCH * NG, 128, 0, stream>>>(center, pos1_w, pos1_b, p128);
  gemm(p128, w_pos2, pos2_b, nullptr, 0, nullptr, nullptr, nullptr, post, nullptr,
       BATCH * NG, CDIM, 128, 0);
  assemble_kernel<<<TOKTOT, CDIM, 0, stream>>>(tok, post, cls_tok, cls_pos, xbuf, posb);

  // ---- transformer ----
  for (int l = 0; l < DEPTH; l++) {
    const bf16* wq = w_qkv  + (size_t)l * 1152 * CDIM;
    const bf16* wp = w_proj + (size_t)l * CDIM * CDIM;
    const bf16* w1 = w_fc1  + (size_t)l * 1536 * CDIM;
    const bf16* w2 = w_fc2  + (size_t)l * CDIM * 1536;
    ln_kernel<<<TOKTOT, 128, 0, stream>>>(xbuf, posb, n1_g + l * CDIM, n1_b + l * CDIM,
                                          hnb, nullptr);           // x=h=x+pos; hn->bf16
    gemm(hnb, wq, nullptr, nullptr, 0, nullptr, nullptr, nullptr, nullptr, qkvb,
         TOKTOT, 1152, CDIM, 0);                                   // qkv (no bias)
    attn_kernel<<<dim3(33, NHEAD, BATCH), 128, 0, stream>>>(qkvb, attb);
    gemm(attb, wp, proj_b + l * CDIM, nullptr, 0, nullptr, nullptr, xbuf, xbuf, nullptr,
         TOKTOT, CDIM, CDIM, 0);                                   // h += proj(o)
    ln_kernel<<<TOKTOT, 128, 0, stream>>>(xbuf, nullptr, n2_g + l * CDIM, n2_b + l * CDIM,
                                          hnb, nullptr);
    gemm(hnb, w1, fc1_b + l * 1536, nullptr, 0, nullptr, nullptr, nullptr, nullptr, mbuf,
         TOKTOT, 1536, CDIM, 2);                                   // gelu
    gemm(mbuf, w2, fc2_b + l * CDIM, nullptr, 0, nullptr, nullptr, xbuf, xbuf, nullptr,
         TOKTOT, CDIM, 1536, 0);                                   // h += mlp
  }
  ln_kernel<<<TOKTOT, 128, 0, stream>>>(xbuf, nullptr, nf_g, nf_b, nullptr, xn);
  head_kernel<<<BATCH, 256, 0, stream>>>(xn, h1_w, h1_b, h2_w, h2_b, (float*)d_out);
}
